// AttentionMechanismVaswani_47502338294491
// MI455X (gfx1250) — matrix operand order repaired and checked by reading, not compiled, not run on hardware
//
#include <hip/hip_runtime.h>
#include <hip/hip_bf16.h>
#include <stdint.h>

typedef unsigned short u16;
typedef __attribute__((ext_vector_type(16))) __bf16 v16bf;
typedef __attribute__((ext_vector_type(8)))  float  v8f;
typedef __attribute__((ext_vector_type(4)))  float  v4f;
typedef __attribute__((ext_vector_type(8)))  u16    v8u;

#define GG 1024
#define NH 8
#define NB 16
#define DKD 64
#define DI 256
#define NEGINF (-1e30f)

__device__ __forceinline__ u16 f2bf(float f) {
  uint32_t u = __builtin_bit_cast(uint32_t, f);
  u += 0x7FFFu + ((u >> 16) & 1u);
  return (u16)(u >> 16);
}
__device__ __forceinline__ float bf2f(u16 b) {
  uint32_t u = ((uint32_t)b) << 16;
  return __builtin_bit_cast(float, u);
}

union U16v { u16 s[16]; v16bf v; };
struct P2 { v8u a, b; };

__device__ __forceinline__ v16bf ldpair(const u16* p0, const u16* p1) {
  P2 t;
  t.a = *(const v8u*)p0;
  t.b = *(const v8u*)p1;
  return __builtin_bit_cast(v16bf, t);
}
// A-operand (16x32 bf16): lane row = l%16; elems 0-7 = K[(l/16)*8 ..], elems 8-15 = K[16+(l/16)*8 ..]
__device__ __forceinline__ v16bf ld_a(const u16* base, int half) {
  return ldpair(base + half * 8, base + 16 + half * 8);
}
// B-operand (32x16 bf16): lane col = l%16; elems 0-7 = K[(l/16)*8 ..], elems 8-15 = K[16+(l/16)*8 ..]
__device__ __forceinline__ v16bf ld_b(const u16* base, int half) {
  return ldpair(base + half * 8, base + 16 + half * 8);
}

#define WMMA_BF16(A, B, C) \
  __builtin_amdgcn_wmma_f32_16x16x32_bf16(false, (A), false, (B), (short)0, (C), false, false)

// Async global->LDS 16B copy per lane (CDNA5 GLOBAL_LOAD_ASYNC_TO_LDS_B128, ASYNCcnt-tracked).
__device__ __forceinline__ void async_copy16(u16* lds_dst, const u16* gsrc) {
  uint32_t lds_off = (uint32_t)(uintptr_t)lds_dst;  // low 32 bits of generic ptr = LDS byte offset
  asm volatile("global_load_async_to_lds_b128 %0, %1, off"
               :: "v"(lds_off), "v"(gsrc)
               : "memory");
}

// ---------------- Stage 0: split W_[QKV] fp32 -> bf16 hi/lo, transposed to [h][n][k] ----------------
__global__ __launch_bounds__(256) void wsplit_kernel(
    const float* __restrict__ Wq, const float* __restrict__ Wk, const float* __restrict__ Wv,
    u16* __restrict__ Wqh, u16* __restrict__ Wql,
    u16* __restrict__ Wkh, u16* __restrict__ Wkl,
    u16* __restrict__ Wvh, u16* __restrict__ Wvl)
{
  int idx = blockIdx.x * blockDim.x + threadIdx.x;  // [h][n][k]
  if (idx >= NH * DKD * DI) return;
  int k  = idx & (DI - 1);
  int n  = (idx >> 8) & (DKD - 1);
  int hh = idx >> 14;
  int src = (hh * DI + k) * DKD + n;
  float q = Wq[src], kk = Wk[src], v = Wv[src];
  u16 qh = f2bf(q);  Wqh[idx] = qh; Wql[idx] = f2bf(q  - bf2f(qh));
  u16 kh = f2bf(kk); Wkh[idx] = kh; Wkl[idx] = f2bf(kk - bf2f(kh));
  u16 vh = f2bf(v);  Wvh[idx] = vh; Wvl[idx] = f2bf(v  - bf2f(vh));
}

// ---------------- Stage 1: QKV projection via 3-term bf16 WMMA ----------------
__global__ __launch_bounds__(128) void qkv_kernel(
    const float* __restrict__ hsrc,
    const u16* __restrict__ Wqh, const u16* __restrict__ Wql,
    const u16* __restrict__ Wkh, const u16* __restrict__ Wkl,
    const u16* __restrict__ Wvh, const u16* __restrict__ Wvl,
    u16* __restrict__ Qh, u16* __restrict__ Ql,
    u16* __restrict__ Kh, u16* __restrict__ Kl,
    u16* __restrict__ Vt)
{
  const int wave = threadIdx.x >> 5, lane = threadIdx.x & 31;
  const int row = lane & 15, half = lane >> 4;
  int task = blockIdx.x * 4 + wave;         // NB*NH*64 tasks
  int mt = task & 63, hh = (task >> 6) & 7, b = task >> 9;
  int m0 = mt * 16;

  // Build A operands (h rows) split hi/lo: 8 K-chunks of 32
  v16bf Ah[8], Al[8];
  const float* rp = hsrc + ((size_t)b * GG + m0 + row) * DI;
  #pragma unroll
  for (int kc = 0; kc < 8; kc++) {
    int base = kc * 32 + half * 8;
    v4f x0 = *(const v4f*)(rp + base);
    v4f x1 = *(const v4f*)(rp + base + 4);
    v4f x2 = *(const v4f*)(rp + base + 16);
    v4f x3 = *(const v4f*)(rp + base + 20);
    float f[16];
    #pragma unroll
    for (int i = 0; i < 4; i++) { f[i] = x0[i]; f[4 + i] = x1[i]; f[8 + i] = x2[i]; f[12 + i] = x3[i]; }
    U16v uh, ul;
    #pragma unroll
    for (int i = 0; i < 16; i++) {
      u16 hb = f2bf(f[i]);
      uh.s[i] = hb;
      ul.s[i] = f2bf(f[i] - bf2f(hb));
    }
    Ah[kc] = uh.v; Al[kc] = ul.v;
  }

  const size_t bh = (size_t)(b * NH + hh);
  #pragma unroll 1
  for (int mat = 0; mat < 3; mat++) {
    const u16* Wh = (mat == 0) ? Wqh : (mat == 1) ? Wkh : Wvh;
    const u16* Wl = (mat == 0) ? Wql : (mat == 1) ? Wkl : Wvl;
    #pragma unroll 1
    for (int nt = 0; nt < 4; nt++) {
      v8f acc = {};
      const u16* wb = Wh + ((size_t)hh * DKD + nt * 16 + row) * DI;
      const u16* wl = Wl + ((size_t)hh * DKD + nt * 16 + row) * DI;
      #pragma unroll
      for (int kc = 0; kc < 8; kc++) {
        v16bf Bh = ld_b(wb + kc * 32, half);
        v16bf Bl = ld_b(wl + kc * 32, half);
        acc = WMMA_BF16(Ah[kc], Bh, acc);
        acc = WMMA_BF16(Ah[kc], Bl, acc);
        acc = WMMA_BF16(Al[kc], Bh, acc);
      }
      if (mat < 2) {
        u16* oh = ((mat == 0) ? Qh : Kh) + (bh * GG + m0) * DKD;
        u16* ol = ((mat == 0) ? Ql : Kl) + (bh * GG + m0) * DKD;
        #pragma unroll
        for (int r = 0; r < 8; r++) {
          int m = r + 8 * half;
          float x = acc[r];
          u16 hb = f2bf(x);
          oh[(size_t)m * DKD + nt * 16 + row] = hb;
          ol[(size_t)m * DKD + nt * 16 + row] = f2bf(x - bf2f(hb));
        }
      } else {
        // transposed store: Vt[bh][n][m]; per-lane 8 contiguous m-values -> one b128 store
        u16* ov = Vt + (bh * DKD + nt * 16 + row) * GG + m0 + 8 * half;
        v8u pack;
        #pragma unroll
        for (int r = 0; r < 8; r++) pack[r] = f2bf(acc[r]);
        *(v8u*)ov = pack;
      }
    }
  }
}

// Prefetch one 32-key step's K-hi/K-lo/V tiles into LDS (6 async b128 per thread; 12 KB per block).
__device__ __forceinline__ void prefetch_tiles(int tid, int k0,
    const u16* __restrict__ Kph, const u16* __restrict__ Kpl, const u16* __restrict__ Vp,
    u16* dKh, u16* dKl, u16* dV)
{
  #pragma unroll
  for (int p = 0; p < 2; p++) {
    int e = tid * 8 + p * 1024;                 // element offset within 2048-elem tile
    // K tiles are contiguous in global: rows k0..k0+31 of [G,64]
    async_copy16(dKh + e, Kph + (size_t)k0 * DKD + e);
    async_copy16(dKl + e, Kpl + (size_t)k0 * DKD + e);
    // V tile: 64 rows (n) x 32 keys from Vt[n][k0..]
    int n = e >> 5, kk = e & 31;
    async_copy16(dV + e, Vp + (size_t)n * GG + k0 + kk);
  }
}

// ---------------- Stage 2: flash attention; block = 4 waves sharing one (b,head) ----------------
__global__ __launch_bounds__(128) void attn_kernel(
    const u16* __restrict__ Qhp_, const u16* __restrict__ Qlp_,
    const u16* __restrict__ Khp_, const u16* __restrict__ Klp_,
    const u16* __restrict__ Vt,  const int* __restrict__ maskp,
    float* __restrict__ outp)
{
  __shared__ __align__(16) u16 sKh[2][32 * DKD];  // 2 x 4 KB
  __shared__ __align__(16) u16 sKl[2][32 * DKD];  // 2 x 4 KB
  __shared__ __align__(16) u16 sV [2][DKD * 32];  // 2 x 4 KB
  __shared__ __align__(16) u16 Pl[4][16 * 32];    // 4 KB (per-wave P transpose)

  const int tid = threadIdx.x;
  const int wave = tid >> 5, lane = tid & 31;
  const int row = lane & 15, half = lane >> 4;
  int task = blockIdx.x * 4 + wave;
  int qt = task & 63, hh = (task >> 6) & 7, b = task >> 9;
  int m0 = qt * 16;
  const size_t bh = (size_t)(b * NH + hh);

  const u16* qhp = Qhp_ + (bh * GG + m0 + row) * DKD;
  const u16* qlp = Qlp_ + (bh * GG + m0 + row) * DKD;
  v16bf Aqh0 = ld_a(qhp,      half);   // dk 0..31 (hi)
  v16bf Aqh1 = ld_a(qhp + 32, half);   // dk 32..63 (hi)
  v16bf Aql0 = ld_a(qlp,      half);
  v16bf Aql1 = ld_a(qlp + 32, half);

  const u16* Kph = Khp_ + bh * GG * DKD;
  const u16* Kpl = Klp_ + bh * GG * DKD;
  const u16* Vp  = Vt   + bh * DKD * GG;
  const int* mp  = maskp + (size_t)(m0 + 8 * half) * GG;  // row for reg r is +r*GG

  float mrow[8], lrow[8];
  v8f O0 = {}, O1 = {}, O2 = {}, O3 = {};
  #pragma unroll
  for (int r = 0; r < 8; r++) { mrow[r] = NEGINF; lrow[r] = 0.f; }
  const float scale = 0.125f;  // 1/sqrt(64)
  u16* pl = Pl[wave];

  // prologue: async-prefetch step 0 into buffer 0
  prefetch_tiles(tid, 0, Kph, Kpl, Vp, sKh[0], sKl[0], sV[0]);

  for (int j = 0; j < 32; j++) {
    int cur = j & 1;
    int k0 = j * 32;
    if (j + 1 < 32) {
      // issue next step's copies, then drain only the current step's 6 asyncs
      prefetch_tiles(tid, k0 + 32, Kph, Kpl, Vp, sKh[cur ^ 1], sKl[cur ^ 1], sV[cur ^ 1]);
      asm volatile("s_wait_asynccnt 0x6" ::: "memory");
    } else {
      asm volatile("s_wait_asynccnt 0x0" ::: "memory");
    }
    __syncthreads();  // all waves' tiles resident in LDS

    const u16* skh = sKh[cur];
    const u16* skl = sKl[cur];
    const u16* sv  = sV[cur];

    v8f S0 = {}, S1 = {};
    {  // subtile 0: keys k0..k0+15
      const u16* kb = skh + (size_t)row * DKD;
      const u16* kl = skl + (size_t)row * DKD;
      v16bf B0h = ld_b(kb,      half);
      v16bf B1h = ld_b(kb + 32, half);
      v16bf B0l = ld_b(kl,      half);
      v16bf B1l = ld_b(kl + 32, half);
      S0 = WMMA_BF16(Aqh0, B0h, S0);
      S0 = WMMA_BF16(Aqh1, B1h, S0);
      S0 = WMMA_BF16(Aqh0, B0l, S0);
      S0 = WMMA_BF16(Aqh1, B1l, S0);
      S0 = WMMA_BF16(Aql0, B0h, S0);
      S0 = WMMA_BF16(Aql1, B1h, S0);
    }
    {  // subtile 1: keys k0+16..k0+31
      const u16* kb = skh + (size_t)(16 + row) * DKD;
      const u16* kl = skl + (size_t)(16 + row) * DKD;
      v16bf B0h = ld_b(kb,      half);
      v16bf B1h = ld_b(kb + 32, half);
      v16bf B0l = ld_b(kl,      half);
      v16bf B1l = ld_b(kl + 32, half);
      S1 = WMMA_BF16(Aqh0, B0h, S1);
      S1 = WMMA_BF16(Aqh1, B1h, S1);
      S1 = WMMA_BF16(Aqh0, B0l, S1);
      S1 = WMMA_BF16(Aqh1, B1l, S1);
      S1 = WMMA_BF16(Aql0, B0h, S1);
      S1 = WMMA_BF16(Aql1, B1h, S1);
    }
    // scale + mask (mask!=0 -> disallowed)
    #pragma unroll
    for (int r = 0; r < 8; r++) {
      int a0 = mp[r * GG + k0 + row];
      int a1 = mp[r * GG + k0 + 16 + row];
      S0[r] = (a0 == 0) ? S0[r] * scale : NEGINF;
      S1[r] = (a1 == 0) ? S1[r] * scale : NEGINF;
    }
    // running row max (reduce over 16 lanes of each half-group)
    float t[8];
    #pragma unroll
    for (int r = 0; r < 8; r++) t[r] = fmaxf(S0[r], S1[r]);
    #pragma unroll
    for (int xm = 1; xm < 16; xm <<= 1) {
      #pragma unroll
      for (int r = 0; r < 8; r++) t[r] = fmaxf(t[r], __shfl_xor(t[r], xm, 32));
    }
    float alpha[8];
    #pragma unroll
    for (int r = 0; r < 8; r++) {
      float mn = fmaxf(mrow[r], t[r]);
      alpha[r] = __expf(mrow[r] - mn);
      mrow[r] = mn;
    }
    // P = exp(s - m) with masked entries forced to exactly 0
    #pragma unroll
    for (int r = 0; r < 8; r++) {
      S0[r] = (S0[r] <= -1e29f) ? 0.f : __expf(S0[r] - mrow[r]);
      S1[r] = (S1[r] <= -1e29f) ? 0.f : __expf(S1[r] - mrow[r]);
    }
    // running row sum
    #pragma unroll
    for (int r = 0; r < 8; r++) t[r] = S0[r] + S1[r];
    #pragma unroll
    for (int xm = 1; xm < 16; xm <<= 1) {
      #pragma unroll
      for (int r = 0; r < 8; r++) t[r] += __shfl_xor(t[r], xm, 32);
    }
    #pragma unroll
    for (int r = 0; r < 8; r++) lrow[r] = lrow[r] * alpha[r] + t[r];
    // rescale accumulated O
    #pragma unroll
    for (int r = 0; r < 8; r++) {
      O0[r] *= alpha[r]; O1[r] *= alpha[r]; O2[r] *= alpha[r]; O3[r] *= alpha[r];
    }
    // C-layout P -> A-layout via per-wave LDS round trip (DS in-order within a wave)
    #pragma unroll
    for (int r = 0; r < 8; r++) {
      int m = r + 8 * half;
      pl[m * 32 + row]      = f2bf(S0[r]);
      pl[m * 32 + 16 + row] = f2bf(S1[r]);
    }
    __builtin_amdgcn_wave_barrier();
    v16bf Pa = ld_a(pl + row * 32, half);
    __builtin_amdgcn_wave_barrier();
    // O += P(16x32) x V(32x64); B-operand K-contiguous from staged sV[n][key]
    {
      const u16* vb = sv + (size_t)row * 32;
      O0 = WMMA_BF16(Pa, ld_b(vb,           half), O0);
      O1 = WMMA_BF16(Pa, ld_b(vb + 16 * 32, half), O1);
      O2 = WMMA_BF16(Pa, ld_b(vb + 32 * 32, half), O2);
      O3 = WMMA_BF16(Pa, ld_b(vb + 48 * 32, half), O3);
    }
    __syncthreads();  // protect buf[cur] before next iteration's prefetch reuses it
  }

  // epilogue: O / l (fully-masked rows -> 0, matching reference)
  float* op = outp + (bh * GG + m0) * DKD;
  #pragma unroll
  for (int r = 0; r < 8; r++) {
    float inv = (lrow[r] > 0.f) ? 1.f / lrow[r] : 0.f;
    int m = r + 8 * half;
    op[(size_t)m * DKD + row]      = O0[r] * inv;
    op[(size_t)m * DKD + 16 + row] = O1[r] * inv;
    op[(size_t)m * DKD + 32 + row] = O2[r] * inv;
    op[(size_t)m * DKD + 48 + row] = O3[r] * inv;
  }
}

extern "C" void kernel_launch(void* const* d_in, const int* in_sizes, int n_in,
                              void* d_out, int out_size, void* d_ws, size_t ws_size,
                              hipStream_t stream)
{
  const float* h    = (const float*)d_in[0];
  const int*   mask = (const int*)  d_in[1];
  const float* Wq   = (const float*)d_in[2];
  const float* Wk   = (const float*)d_in[3];
  const float* Wv   = (const float*)d_in[4];

  char* ws = (char*)d_ws;
  const size_t WSZ = (size_t)NH * DKD * DI * sizeof(u16);        // 256 KB per weight split
  const size_t QSZ = (size_t)NB * NH * GG * DKD * sizeof(u16);   // 16 MB per Q/K/V buffer
  u16* Wqh = (u16*)(ws);            u16* Wql = (u16*)(ws + 1 * WSZ);
  u16* Wkh = (u16*)(ws + 2 * WSZ);  u16* Wkl = (u16*)(ws + 3 * WSZ);
  u16* Wvh = (u16*)(ws + 4 * WSZ);  u16* Wvl = (u16*)(ws + 5 * WSZ);
  char* p = ws + 6 * WSZ;
  u16* Qh = (u16*)(p);            u16* Ql = (u16*)(p + 1 * QSZ);
  u16* Kh = (u16*)(p + 2 * QSZ);  u16* Kl = (u16*)(p + 3 * QSZ);
  u16* Vt = (u16*)(p + 4 * QSZ);

  wsplit_kernel<<<(NH * DKD * DI + 255) / 256, 256, 0, stream>>>(
      Wq, Wk, Wv, Wqh, Wql, Wkh, Wkl, Wvh, Wvl);
  qkv_kernel<<<NB * NH * 64 / 4, 128, 0, stream>>>(
      h, Wqh, Wql, Wkh, Wkl, Wvh, Wvl, Qh, Ql, Kh, Kl, Vt);
  attn_kernel<<<NB * NH * 64 / 4, 128, 0, stream>>>(
      Qh, Ql, Kh, Kl, Vt, mask, (float*)d_out);

  (void)in_sizes; (void)n_in; (void)out_size; (void)ws_size;
}
